// AttentionalGNN_44238163149253
// MI455X (gfx1250) — matrix operand order repaired and checked by reading, not compiled, not run on hardware
//
#include <hip/hip_runtime.h>
#include <hip/hip_bf16.h>

// ---------------------------------------------------------------------------
// Types for CDNA5 WMMA (wave32, V_WMMA_F32_16X16X32_BF16)
// ---------------------------------------------------------------------------
typedef __attribute__((ext_vector_type(16))) __bf16 v16bf;
typedef __attribute__((ext_vector_type(8)))  __bf16 v8bf;
typedef __attribute__((ext_vector_type(8)))  float  v8f;

#define D_DIM 256
#define N_PTS 1024
#define NHEAD 4
#define BLK_M 64
#define BLK_N 128
#define BLK_K 32

__device__ __forceinline__ __bf16 f2bf(float f) {
  // round-to-nearest-even fp32 -> bf16
  unsigned u = __builtin_bit_cast(unsigned, f);
  unsigned r = u + 0x7FFFu + ((u >> 16) & 1u);
  unsigned short h = (unsigned short)(r >> 16);
  return __builtin_bit_cast(__bf16, h);
}

// A fragment (16x32 bf16, M x K). ISA layout: lanes 0-15 row M=lane, elems
// 0..7 -> K 0..7, elems 8..15 -> K 16..23 ; lanes 16-31 same rows, +8 K offset.
// With LDS rows of 32 bf16 this is two aligned 16B loads per lane.
__device__ __forceinline__ v16bf frag_a(const __bf16* As, int row, int lane) {
  const __bf16* p = As + (row + (lane & 15)) * BLK_K + (lane >> 4) * 8;
  v8bf lo = *(const v8bf*)p;
  v8bf hi = *(const v8bf*)(p + 16);
  return __builtin_shufflevector(lo, hi, 0,1,2,3,4,5,6,7,8,9,10,11,12,13,14,15);
}

// B fragment (32x16 bf16, K x N) from a K-contiguous transposed LDS tile
// Bt[n][k]: lane n = lane&15, elems 0..7 -> K 0..7, 8..15 -> K 16..23; lanes 16-31 +8 K.
__device__ __forceinline__ v16bf frag_b(const __bf16* Bs, int col, int lane) {
  const __bf16* p = Bs + (col + (lane & 15)) * BLK_K + (lane >> 4) * 8;
  v8bf lo = *(const v8bf*)p;
  v8bf hi = *(const v8bf*)(p + 16);
  return __builtin_shufflevector(lo, hi, 0,1,2,3,4,5,6,7,8,9,10,11,12,13,14,15);
}

__device__ __forceinline__ v8f wmma_bf16(v16bf a, v16bf b, v8f c) {
  return __builtin_amdgcn_wmma_f32_16x16x32_bf16(false, a, false, b, (short)0, c,
                                                 false, false);
}

// ---------------------------------------------------------------------------
// Generic weight GEMM: C[M,N] = W[M,K] @ X[K,N] + bias, optional ReLU.
// X rows < Ksplit come from X1, rows >= Ksplit from X2 (fused concat).
// ---------------------------------------------------------------------------
__global__ __launch_bounds__(256) void gemm_wx(
    const float* __restrict__ W, const float* __restrict__ X1,
    const float* __restrict__ X2, const float* __restrict__ bias,
    float* __restrict__ C, int M, int N, int K, int Ksplit, int relu)
{
  __shared__ __attribute__((aligned(16))) __bf16 As[BLK_M * BLK_K];
  __shared__ __attribute__((aligned(16))) __bf16 Bs[BLK_N * BLK_K];
  const int t = threadIdx.x;
  const int lane = t & 31, w = t >> 5;
  const int wm = w >> 2, wn = w & 3;           // 2 x 4 wave grid
  const int M0 = blockIdx.y * BLK_M, N0 = blockIdx.x * BLK_N;

  v8f acc[2][2] = {};
  for (int k0 = 0; k0 < K; k0 += BLK_K) {
#pragma unroll
    for (int r = 0; r < 8; ++r) {              // 64x32 A tile
      int idx = t + r * 256;
      int j = idx & 31, i = idx >> 5;
      As[i * BLK_K + j] = f2bf(W[(size_t)(M0 + i) * K + k0 + j]);
    }
#pragma unroll
    for (int r = 0; r < 16; ++r) {             // 128x32 B tile, transposed
      int idx = t + r * 256;
      int n = idx & 127, kk = idx >> 7;
      int kg = k0 + kk;
      const float* Xp = (kg < Ksplit) ? (X1 + (size_t)kg * N)
                                      : (X2 + (size_t)(kg - Ksplit) * N);
      Bs[n * BLK_K + kk] = f2bf(Xp[N0 + n]);
    }
    __syncthreads();
#pragma unroll
    for (int i = 0; i < 2; ++i) {
      v16bf a = frag_a(As, wm * 32 + i * 16, lane);
#pragma unroll
      for (int j = 0; j < 2; ++j) {
        v16bf b = frag_b(Bs, wn * 32 + j * 16, lane);
        acc[i][j] = wmma_bf16(a, b, acc[i][j]);
      }
    }
    __syncthreads();
  }

  const int nloc = lane & 15, hi = lane >> 4;
#pragma unroll
  for (int i = 0; i < 2; ++i)
#pragma unroll
    for (int j = 0; j < 2; ++j) {
      int col = N0 + wn * 32 + j * 16 + nloc;
#pragma unroll
      for (int e = 0; e < 8; ++e) {
        int row = M0 + wm * 32 + i * 16 + hi * 8 + e;
        float v = acc[i][j][e] + bias[row];
        if (relu) v = fmaxf(v, 0.f);
        C[(size_t)row * N + col] = v;
      }
    }
}

// ---------------------------------------------------------------------------
// Attention scores: S[h,n,m] = scale * sum_d Q[4d+h, n] * K[4d+h, m]
// (heads interleaved in channel dim per jnp reshape [dh, H]).
// GEMM M-dim = query n (64 tile), N-dim = key m (128 tile), K = 64.
// ---------------------------------------------------------------------------
__global__ __launch_bounds__(256) void attn_scores(
    const float* __restrict__ Q, const float* __restrict__ Kmat,
    float* __restrict__ S, float scale)
{
  __shared__ __attribute__((aligned(16))) __bf16 As[BLK_M * BLK_K];
  __shared__ __attribute__((aligned(16))) __bf16 Bs[BLK_N * BLK_K];
  const int t = threadIdx.x;
  const int lane = t & 31, w = t >> 5;
  const int wm = w >> 2, wn = w & 3;
  const int h = blockIdx.z;
  const int n0 = blockIdx.y * BLK_M;
  const int m0 = blockIdx.x * BLK_N;

  v8f acc[2][2] = {};
#pragma unroll
  for (int k0 = 0; k0 < 64; k0 += BLK_K) {
#pragma unroll
    for (int r = 0; r < 8; ++r) {              // As[n][d] = Q[4(k0+d)+h][n0+n]
      int idx = t + r * 256;
      int i = idx & 63, j = idx >> 6;
      As[i * BLK_K + j] = f2bf(Q[(size_t)((k0 + j) * NHEAD + h) * N_PTS + n0 + i]);
    }
#pragma unroll
    for (int r = 0; r < 16; ++r) {             // Bs[m][d] = K[4(k0+d)+h][m0+m]
      int idx = t + r * 256;
      int m = idx & 127, j = idx >> 7;
      Bs[m * BLK_K + j] = f2bf(Kmat[(size_t)((k0 + j) * NHEAD + h) * N_PTS + m0 + m]);
    }
    __syncthreads();
#pragma unroll
    for (int i = 0; i < 2; ++i) {
      v16bf a = frag_a(As, wm * 32 + i * 16, lane);
#pragma unroll
      for (int j = 0; j < 2; ++j) {
        v16bf b = frag_b(Bs, wn * 32 + j * 16, lane);
        acc[i][j] = wmma_bf16(a, b, acc[i][j]);
      }
    }
    __syncthreads();
  }

  const int nloc = lane & 15, hi = lane >> 4;
#pragma unroll
  for (int i = 0; i < 2; ++i)
#pragma unroll
    for (int j = 0; j < 2; ++j) {
      int m = m0 + wn * 32 + j * 16 + nloc;
#pragma unroll
      for (int e = 0; e < 8; ++e) {
        int n = n0 + wm * 32 + i * 16 + hi * 8 + e;
        S[((size_t)h * N_PTS + n) * N_PTS + m] = acc[i][j][e] * scale;
      }
    }
}

// ---------------------------------------------------------------------------
// Attention apply: X[4d+h, n] = sum_m P[h,n,m] * V[4d+h, m]
// GEMM M=64 (d), N-dim = n (128 tile), K = 1024 (m). P rows are K-contiguous,
// i.e. already in the transposed-B layout the fragment loader wants.
// ---------------------------------------------------------------------------
__global__ __launch_bounds__(256) void attn_apply(
    const float* __restrict__ V, const float* __restrict__ P,
    float* __restrict__ Xo)
{
  __shared__ __attribute__((aligned(16))) __bf16 As[BLK_M * BLK_K];
  __shared__ __attribute__((aligned(16))) __bf16 Bs[BLK_N * BLK_K];
  const int t = threadIdx.x;
  const int lane = t & 31, w = t >> 5;
  const int wm = w >> 2, wn = w & 3;
  const int h = blockIdx.z;
  const int n0 = blockIdx.x * BLK_N;

  v8f acc[2][2] = {};
  for (int k0 = 0; k0 < N_PTS; k0 += BLK_K) {
#pragma unroll
    for (int r = 0; r < 8; ++r) {              // As[d][m] = V[4d+h][k0+m]
      int idx = t + r * 256;
      int j = idx & 31, i = idx >> 5;
      As[i * BLK_K + j] = f2bf(V[(size_t)(i * NHEAD + h) * N_PTS + k0 + j]);
    }
#pragma unroll
    for (int r = 0; r < 16; ++r) {             // Bs[n][m] = P[h][n0+n][k0+m]
      int idx = t + r * 256;
      int j = idx & 31, n = idx >> 5;
      Bs[n * BLK_K + j] = f2bf(P[((size_t)h * N_PTS + n0 + n) * N_PTS + k0 + j]);
    }
    __syncthreads();
#pragma unroll
    for (int i = 0; i < 2; ++i) {
      v16bf a = frag_a(As, wm * 32 + i * 16, lane);
#pragma unroll
      for (int j = 0; j < 2; ++j) {
        v16bf b = frag_b(Bs, wn * 32 + j * 16, lane);
        acc[i][j] = wmma_bf16(a, b, acc[i][j]);
      }
    }
    __syncthreads();
  }

  const int nloc = lane & 15, hi = lane >> 4;
#pragma unroll
  for (int i = 0; i < 2; ++i)
#pragma unroll
    for (int j = 0; j < 2; ++j) {
      int col = n0 + wn * 32 + j * 16 + nloc;
#pragma unroll
      for (int e = 0; e < 8; ++e) {
        int d = wm * 32 + i * 16 + hi * 8 + e;     // 0..63
        Xo[(size_t)(d * NHEAD + h) * N_PTS + col] = acc[i][j][e];
      }
    }
}

// ---------------------------------------------------------------------------
// In-place softmax over last dim of S[4][1024][1024]; one block per row.
// ---------------------------------------------------------------------------
__global__ __launch_bounds__(256) void softmax_rows(float* __restrict__ S)
{
  float* p = S + (size_t)blockIdx.x * N_PTS;
  __shared__ float red[256];
  const int t = threadIdx.x;
  float vals[4];
  float mx = -3.4e38f;
#pragma unroll
  for (int r = 0; r < 4; ++r) { vals[r] = p[t + r * 256]; mx = fmaxf(mx, vals[r]); }
  red[t] = mx; __syncthreads();
  for (int s = 128; s > 0; s >>= 1) { if (t < s) red[t] = fmaxf(red[t], red[t + s]); __syncthreads(); }
  mx = red[0]; __syncthreads();
  float sum = 0.f;
#pragma unroll
  for (int r = 0; r < 4; ++r) { vals[r] = __expf(vals[r] - mx); sum += vals[r]; }
  red[t] = sum; __syncthreads();
  for (int s = 128; s > 0; s >>= 1) { if (t < s) red[t] += red[t + s]; __syncthreads(); }
  float inv = 1.0f / red[0];
#pragma unroll
  for (int r = 0; r < 4; ++r) p[t + r * 256] = vals[r] * inv;
}

// ---------------------------------------------------------------------------
// out = LN(a (+ b) (+ c)) over channel dim (256), per position n.
// One block (256 threads = D) per position; safe in-place (out may alias a).
// ---------------------------------------------------------------------------
__global__ __launch_bounds__(256) void layernorm_sum(
    const float* __restrict__ a, const float* __restrict__ b,
    const float* __restrict__ c, const float* __restrict__ g,
    const float* __restrict__ beta, float* __restrict__ out)
{
  const int n = blockIdx.x;
  const int d = threadIdx.x;
  const size_t o = (size_t)d * N_PTS + n;
  float x = a[o];
  if (b) x += b[o];
  if (c) x += c[o];
  __shared__ float red[256];
  red[d] = x; __syncthreads();
  for (int s = 128; s > 0; s >>= 1) { if (d < s) red[d] += red[d + s]; __syncthreads(); }
  float mu = red[0] * (1.0f / D_DIM);
  __syncthreads();
  float dv = x - mu;
  red[d] = dv * dv; __syncthreads();
  for (int s = 128; s > 0; s >>= 1) { if (d < s) red[d] += red[d + s]; __syncthreads(); }
  float var = red[0] * (1.0f / D_DIM);
  out[o] = dv * rsqrtf(var + 1e-5f) * g[d] + beta[d];
}

// ---------------------------------------------------------------------------
// Final output: out[m] = any_i(unreach[i][m]) ? -1e9 : mean_{h,n} S[h,n,m]
// (raw pre-softmax logits, per the reference's returned weights).
// ---------------------------------------------------------------------------
__global__ void score_out(const float* __restrict__ S,
                          const unsigned char* __restrict__ unreach,
                          float* __restrict__ out)
{
  const int m = blockIdx.x * blockDim.x + threadIdx.x;
  if (m >= N_PTS) return;
  float s = 0.f;
  for (int r = 0; r < NHEAD * N_PTS; ++r) s += S[(size_t)r * N_PTS + m];
  bool fr = false;
  for (int i = 0; i < N_PTS; ++i) fr = fr || (unreach[(size_t)i * N_PTS + m] != 0);
  out[m] = fr ? -1e9f : s * (1.0f / (NHEAD * N_PTS));
}

// ---------------------------------------------------------------------------
// Host orchestration
// ---------------------------------------------------------------------------
extern "C" void kernel_launch(void* const* d_in, const int* in_sizes, int n_in,
                              void* d_out, int out_size, void* d_ws, size_t ws_size,
                              hipStream_t stream) {
  (void)in_sizes; (void)n_in; (void)out_size; (void)ws_size;
  const int L = 6;
  const size_t SZ = (size_t)D_DIM * N_PTS;          // 262144 floats per activation

  const float* desc[4] = {(const float*)d_in[0], (const float*)d_in[1],
                          (const float*)d_in[2], (const float*)d_in[3]};
  const unsigned char* unreach = (const unsigned char*)d_in[5];
  const float* proj_w  = (const float*)d_in[8];     // [3,L,3,D,D]
  const float* proj_b  = (const float*)d_in[9];     // [3,L,3,D]
  const float* merge_w = (const float*)d_in[10];    // [3,L,D,D]
  const float* merge_b = (const float*)d_in[11];    // [3,L,D]
  const float* mlp_w1  = (const float*)d_in[12];    // [3,L,2D,2D]
  const float* mlp_b1  = (const float*)d_in[13];    // [3,L,2D]
  const float* mlp_w2  = (const float*)d_in[14];    // [3,L,D,2D]
  const float* mlp_b2  = (const float*)d_in[15];    // [3,L,D]
  const float* norm_g  = (const float*)d_in[16];    // [4,L,D]
  const float* norm_b  = (const float*)d_in[17];    // [4,L,D]

  float* ws = (float*)d_ws;
  float* dbuf[4]; for (int i = 0; i < 4; ++i) dbuf[i] = ws + i * SZ;
  float* Qb   = ws + 4  * SZ;
  float* Kb   = ws + 5  * SZ;
  float* Vb   = ws + 6  * SZ;
  float* Xa   = ws + 7  * SZ;
  float* msg  = ws + 8  * SZ;
  float* eb[5]; for (int i = 0; i < 5; ++i) eb[i] = ws + (9 + i) * SZ;
  float* hbuf = ws + 14 * SZ;                       // 512*1024 floats (2*SZ)
  float* S    = ws + 16 * SZ;                       // 4*1024*1024 floats (16*SZ)

  for (int i = 0; i < 4; ++i)
    hipMemcpyAsync(dbuf[i], desc[i], SZ * sizeof(float),
                   hipMemcpyDeviceToDevice, stream);

  auto gemm = [&](const float* W, const float* X1, const float* X2,
                  const float* bias, float* C, int M, int N, int K,
                  int Ksplit, int relu) {
    dim3 grid(N / BLK_N, M / BLK_M);
    gemm_wx<<<grid, 256, 0, stream>>>(W, X1, X2, bias, C, M, N, K, Ksplit, relu);
  };

  auto prop = [&](int br, int i, const float* x, const float* src,
                  float* delta, bool do_score) {
    const float* pw = proj_w + (size_t)((br * L + i) * 3) * D_DIM * D_DIM;
    const float* pb = proj_b + (size_t)((br * L + i) * 3) * D_DIM;
    gemm(pw + 0 * D_DIM * D_DIM, x,   nullptr, pb + 0 * D_DIM, Qb, D_DIM, N_PTS, D_DIM, D_DIM, 0);
    gemm(pw + 1 * D_DIM * D_DIM, src, nullptr, pb + 1 * D_DIM, Kb, D_DIM, N_PTS, D_DIM, D_DIM, 0);
    gemm(pw + 2 * D_DIM * D_DIM, src, nullptr, pb + 2 * D_DIM, Vb, D_DIM, N_PTS, D_DIM, D_DIM, 0);

    attn_scores<<<dim3(N_PTS / BLK_N, N_PTS / BLK_M, NHEAD), 256, 0, stream>>>(
        Qb, Kb, S, 0.125f);  // 1/sqrt(dh=64)
    if (do_score)
      score_out<<<dim3(N_PTS / 256), 256, 0, stream>>>(S, unreach, (float*)d_out);
    softmax_rows<<<NHEAD * N_PTS, 256, 0, stream>>>(S);
    attn_apply<<<dim3(N_PTS / BLK_N, 1, NHEAD), 256, 0, stream>>>(Vb, S, Xa);

    const float* mw = merge_w + (size_t)(br * L + i) * D_DIM * D_DIM;
    const float* mb = merge_b + (size_t)(br * L + i) * D_DIM;
    gemm(mw, Xa, nullptr, mb, msg, D_DIM, N_PTS, D_DIM, D_DIM, 0);

    const float* w1 = mlp_w1 + (size_t)(br * L + i) * (2 * D_DIM) * (2 * D_DIM);
    const float* b1 = mlp_b1 + (size_t)(br * L + i) * (2 * D_DIM);
    gemm(w1, x, msg, b1, hbuf, 2 * D_DIM, N_PTS, 2 * D_DIM, D_DIM, 1);  // concat+relu

    const float* w2 = mlp_w2 + (size_t)(br * L + i) * D_DIM * (2 * D_DIM);
    const float* b2 = mlp_b2 + (size_t)(br * L + i) * D_DIM;
    gemm(w2, hbuf, nullptr, b2, delta, D_DIM, N_PTS, 2 * D_DIM, 2 * D_DIM, 0);
  };

  auto lnorm = [&](float* out, const float* a, const float* b, const float* c,
                   int which, int i) {
    layernorm_sum<<<N_PTS, 256, 0, stream>>>(
        a, b, c, norm_g + (size_t)(which * L + i) * D_DIM,
        norm_b + (size_t)(which * L + i) * D_DIM, out);
  };

  const bool isCross[L] = {false, true, false, true, false, true};
  for (int i = 0; i < L; ++i) {
    const bool cross = isCross[i];
    const float* src0 = cross ? dbuf[1] : dbuf[0];
    const float* src1 = cross ? dbuf[0] : dbuf[1];
    prop(0, i, dbuf[0], src0, eb[0], false);
    prop(0, i, dbuf[1], src1, eb[1], i == L - 1);   // score1 kept from last layer
    if (cross) {
      prop(1, i, dbuf[2], dbuf[1], eb[2], false);   // delta21
      prop(1, i, dbuf[1], dbuf[2], eb[3], false);   // delta12 (pre-update d2)
      lnorm(dbuf[2], dbuf[2], eb[2], nullptr, 2, i);
      prop(2, i, dbuf[0], dbuf[3], eb[2], false);   // delta03 (reuse eb[2])
      prop(2, i, dbuf[3], dbuf[0], eb[4], false);   // delta30 (pre-update d0)
      lnorm(dbuf[3], dbuf[3], eb[4], nullptr, 3, i);
      lnorm(dbuf[0], dbuf[0], eb[0], eb[2], 0, i);  // LN(d0+delta0+delta03)
      lnorm(dbuf[1], dbuf[1], eb[1], eb[3], 1, i);  // LN(d1+delta1+delta12)
    } else {
      prop(1, i, dbuf[2], dbuf[2], eb[2], false);
      lnorm(dbuf[2], dbuf[2], eb[2], nullptr, 2, i);
      prop(2, i, dbuf[3], dbuf[3], eb[3], false);
      lnorm(dbuf[3], dbuf[3], eb[3], nullptr, 3, i);
      lnorm(dbuf[0], dbuf[0], eb[0], nullptr, 0, i);
      lnorm(dbuf[1], dbuf[1], eb[1], nullptr, 1, i);
    }
  }
}
